// SimilarityModel_20581483283017
// MI455X (gfx1250) — matrix operand order repaired and checked by reading, not compiled, not run on hardware
//
#include <hip/hip_runtime.h>
#include <hip/hip_bf16.h>

typedef __attribute__((ext_vector_type(16))) _Float16 v16h;
typedef __attribute__((ext_vector_type(8)))  _Float16 v8h;
typedef __attribute__((ext_vector_type(8)))  float    v8f;
typedef __attribute__((__vector_size__(16))) int      v4i;

static constexpr int NP = 9216;   // 96*96 pixels
static constexpr int ND = 1024;   // channels / embed dim

#if defined(__AMDGCN__) && __has_builtin(__builtin_amdgcn_global_load_async_to_lds_b128)
#define HAVE_ASYNC_LDS 1
#else
#define HAVE_ASYNC_LDS 0
#endif

#if __has_builtin(__builtin_amdgcn_s_wait_asynccnt)
#define WAIT_ASYNC(n) __builtin_amdgcn_s_wait_asynccnt(n)
#else
#define WAIT_ASYNC(n) asm volatile("s_wait_asynccnt %0" ::"n"(n) : "memory")
#endif

// ---------- WMMA fragment helpers (CDNA5 f16 16x16x32 layouts, ISA 7.12.2) ----------
__device__ __forceinline__ v16h frag_cat(v8h lo, v8h hi) {
  return __builtin_shufflevector(lo, hi, 0,1,2,3,4,5,6,7,8,9,10,11,12,13,14,15);
}
// A operand (16x32, M x K) from row-major M[row][k], ld in halves.
// lane&15 -> row; 16 contiguous K at (lane>>4)*16  => two adjacent 16B loads
__device__ __forceinline__ v16h load_frag_a(const _Float16* M, int ld, int row0, int k0, int lane) {
  const _Float16* p = M + (size_t)(row0 + (lane & 15)) * ld + (k0 + ((lane >> 4) << 4));
  return frag_cat(*(const v8h*)p, *(const v8h*)(p + 8));
}
// B operand (32x16, K x N) sourced from a ROW-MAJOR (N x K) matrix (i.e. B^T stored row-major).
// lane&15 -> column n; 16 contiguous K at (lane>>4)*16  => two adjacent 16B loads
__device__ __forceinline__ v16h load_frag_b(const _Float16* M, int ld, int col0, int k0, int lane) {
  const _Float16* p = M + (size_t)(col0 + (lane & 15)) * ld + (k0 + ((lane >> 4) << 4));
  return frag_cat(*(const v8h*)p, *(const v8h*)(p + 8));
}

// ---------- prep: transpose (C,N)->(N,C) f32->f16, add patch embedding for A ----------
__global__ void prep_kernel(const float* __restrict__ rf, const int* __restrict__ sig,
                            const float* __restrict__ pe, _Float16* __restrict__ Af,
                            _Float16* __restrict__ Of) {
  __shared__ float tile[32][33];
  int n0 = blockIdx.x * 32, c0 = blockIdx.y * 32;
  int tx = threadIdx.x, ty = threadIdx.y;
#pragma unroll
  for (int i = 0; i < 4; ++i) {
    int c = c0 + ty + i * 8;
    tile[ty + i * 8][tx] = rf[(size_t)c * NP + (n0 + tx)];
  }
  __syncthreads();
#pragma unroll
  for (int i = 0; i < 4; ++i) {
    int n = n0 + ty + i * 8;
    int c = c0 + tx;
    float v = tile[tx][ty + i * 8];
    Of[(size_t)n * ND + c] = (_Float16)v;
    int s = sig[n];
    Af[(size_t)n * ND + c] = (_Float16)(v + pe[s * ND + c]);
  }
}

// ---------- convert Wq / Wk to f16 (layout unchanged: row-major (out,in)) ----------
__global__ void cvt_w_kernel(const float* __restrict__ Wq, const float* __restrict__ Wk,
                             _Float16* __restrict__ Wqh, _Float16* __restrict__ Wkh) {
  int i = blockIdx.x * blockDim.x + threadIdx.x;
  if (i < ND * ND)            Wqh[i]            = (_Float16)Wq[i];
  else                        Wkh[i - ND * ND]  = (_Float16)Wk[i - ND * ND];
}

// ---------- fold conv into V:  wvc[k] = sum_c cw[c]*Wv[c][k];  bconst = cw.bv ----------
__global__ void wvc_kernel(const float* __restrict__ Wv, const float* __restrict__ cw,
                           const float* __restrict__ bv, float* __restrict__ wvc,
                           float* __restrict__ bconst) {
  int k = blockIdx.x * blockDim.x + threadIdx.x;   // 0..1023
  float a = 0.f;
  for (int c = 0; c < ND; ++c) a += cw[c] * Wv[(size_t)c * ND + k];
  wvc[k] = a;
  if (k == 0) {
    float b = 0.f;
    for (int c = 0; c < ND; ++c) b += cw[c] * bv[c];
    *bconst = b;
  }
}

// ---------- vscal[n] = O_flat[n] . wvc + bconst  (folded V projection + conv) ----------
__global__ void vscal_kernel(const float* __restrict__ rf, const float* __restrict__ wvc,
                             const float* __restrict__ bconst, float* __restrict__ vs) {
  int n = blockIdx.x * blockDim.x + threadIdx.x;
  if (n >= NP) return;
  float a = *bconst;
  for (int k = 0; k < ND; ++k) a += rf[(size_t)k * NP + n] * wvc[k];  // coalesced over n
  vs[n] = a;
}

// ---------- projection GEMM: Y(N x D, f16) = X(N x D, f16) @ W^T + bias ----------
__global__ void __launch_bounds__(128) proj_kernel(const _Float16* __restrict__ X,
                                                   const _Float16* __restrict__ W,
                                                   const float* __restrict__ bias,
                                                   _Float16* __restrict__ Y) {
  int lane = threadIdx.x & 31, wv = threadIdx.x >> 5;
  int row0 = blockIdx.x * 16;
  int col0 = (blockIdx.y * 4 + wv) * 16;
  v8f acc = {};
#pragma unroll 4
  for (int k0 = 0; k0 < ND; k0 += 32) {
    v16h a = load_frag_a(X, ND, row0, k0, lane);
    v16h b = load_frag_b(W, ND, col0, k0, lane);   // W row-major (out,in) == B^T
    acc = __builtin_amdgcn_wmma_f32_16x16x32_f16(false, a, false, b, (short)0, acc, false, false);
  }
  int col = col0 + (lane & 15);
  float bc = bias[col];
  int rbase = row0 + ((lane >> 4) << 3);           // D layout: vgpr r -> row r / r+8
#pragma unroll
  for (int r = 0; r < 8; ++r)
    Y[(size_t)(rbase + r) * ND + col] = (_Float16)(acc[r] + bc);
}

#if HAVE_ASYNC_LDS
// issue one wave's share of a 16x1024-f16 K-tile fill: 16 async b128 per thread-id slot
__device__ __forceinline__ void issue_ktile_async(const _Float16* Km, _Float16* kb,
                                                  int j0, int tid) {
#pragma unroll
  for (int it = 0; it < 16; ++it) {
    int chunk = it * 128 + tid;                    // 2048 chunks of 8 halves (16B)
    int krow = chunk >> 7;
    int kcol = (chunk & 127) << 3;
    __builtin_amdgcn_global_load_async_to_lds_b128(
        (__attribute__((address_space(1))) v4i*)(Km + (size_t)(j0 + krow) * ND + kcol),
        (__attribute__((address_space(3))) v4i*)(kb + krow * ND + kcol),
        0, 0);
  }
}
#endif

// ---------- flash attention: seg[n] = softmax_j(Q.K^T/32)[n] . vscal; out = sigmoid ----------
__global__ void __launch_bounds__(128, 1) attn_kernel(const _Float16* __restrict__ Q,
                                                      const _Float16* __restrict__ Km,
                                                      const float* __restrict__ vs,
                                                      const float* __restrict__ cb,
                                                      float* __restrict__ out) {
#if HAVE_ASYNC_LDS
  __shared__ _Float16 kbuf[2][16 * ND];            // 64 KB double-buffered K tile
#else
  __shared__ _Float16 kbuf[1][16 * ND];            // 32 KB single buffer
#endif
  int tid = threadIdx.x, lane = tid & 31, wv = tid >> 5;
  int row0 = blockIdx.x * 64 + wv * 16;

  // Q tile (16 rows x 1024) held entirely in VGPRs: 32 A-fragments
  v16h qf[32];
#pragma unroll
  for (int kk = 0; kk < 32; ++kk)
    qf[kk] = load_frag_a(Q, ND, row0, kk * 32, lane);

  const float scale = 0.03125f;                    // 1/sqrt(1024)
  float m[8], ssum[8], acc[8];
#pragma unroll
  for (int r = 0; r < 8; ++r) { m[r] = -3.0e38f; ssum[r] = 0.f; acc[r] = 0.f; }

#if HAVE_ASYNC_LDS
  issue_ktile_async(Km, &kbuf[0][0], 0, tid);      // prefetch tile 0
#endif

  for (int jt = 0; jt < NP / 16; ++jt) {           // stream all 9216 keys
    int j0 = jt * 16;
#if HAVE_ASYNC_LDS
    if (jt + 1 < NP / 16) {                        // prefetch next tile, then wait for current
      issue_ktile_async(Km, &kbuf[(jt + 1) & 1][0], j0 + 16, tid);
      WAIT_ASYNC(16);                              // in-order: current tile's 16 are done
    } else {
      WAIT_ASYNC(0);
    }
    __syncthreads();                               // all waves' fills of tile jt visible
    const _Float16* kb = &kbuf[jt & 1][0];
#else
    __syncthreads();
#pragma unroll
    for (int it = 0; it < 16; ++it) {              // synchronous 32KB tile fill
      int chunk = it * 128 + tid;
      int krow = chunk >> 7;
      int kcol = (chunk & 127) << 3;
      *(v8h*)&kbuf[0][krow * ND + kcol] = *(const v8h*)&Km[(size_t)(j0 + krow) * ND + kcol];
    }
    __syncthreads();
    const _Float16* kb = &kbuf[0][0];
#endif

    v8f st = {};
#pragma unroll
    for (int kk = 0; kk < 32; ++kk) {
      const _Float16* p = kb + (lane & 15) * ND + kk * 32 + ((lane >> 4) << 4);
      v16h b = frag_cat(*(const v8h*)p, *(const v8h*)(p + 8));
      st = __builtin_amdgcn_wmma_f32_16x16x32_f16(false, qf[kk], false, b, (short)0, st, false, false);
    }

    float vj = vs[j0 + (lane & 15)];
#pragma unroll
    for (int r = 0; r < 8; ++r) {                  // per-lane online softmax
      float s = st[r] * scale;
      float mn = fmaxf(m[r], s);
      float corr = __expf(m[r] - mn);
      float e = __expf(s - mn);
      ssum[r] = ssum[r] * corr + e;
      acc[r]  = acc[r]  * corr + e * vj;
      m[r] = mn;
    }
#if HAVE_ASYNC_LDS
    __syncthreads();                               // buffer reusable only after all waves read it
#endif
  }

  // merge the 16 column-partials per row (lanes 0-15 hold rows r, 16-31 rows r+8)
#pragma unroll
  for (int r = 0; r < 8; ++r) {
#pragma unroll
    for (int mask = 1; mask < 16; mask <<= 1) {
      float mo = __shfl_xor(m[r], mask, 32);
      float so = __shfl_xor(ssum[r], mask, 32);
      float ao = __shfl_xor(acc[r], mask, 32);
      float mn = fmaxf(m[r], mo);
      float c1 = __expf(m[r] - mn), c2 = __expf(mo - mn);
      ssum[r] = ssum[r] * c1 + so * c2;
      acc[r]  = acc[r]  * c1 + ao * c2;
      m[r] = mn;
    }
  }
  if ((lane & 15) == 0) {
    float cbv = *cb;
    int rb = row0 + ((lane >> 4) << 3);
#pragma unroll
    for (int r = 0; r < 8; ++r) {
      float seg = acc[r] / ssum[r] + cbv;
      out[rb + r] = 1.0f / (1.0f + __expf(-seg));
    }
  }
}

extern "C" void kernel_launch(void* const* d_in, const int* in_sizes, int n_in,
                              void* d_out, int out_size, void* d_ws, size_t ws_size,
                              hipStream_t stream) {
  const int*   sig = (const int*)d_in[0];
  const float* rf  = (const float*)d_in[1];
  // d_in[2] = roi_mask: only its shape is used by the reference -> data unused
  const float* pe  = (const float*)d_in[3];
  const float* Wq  = (const float*)d_in[4];
  const float* bq  = (const float*)d_in[5];
  const float* Wk  = (const float*)d_in[6];
  const float* bk  = (const float*)d_in[7];
  const float* Wv  = (const float*)d_in[8];
  const float* bv  = (const float*)d_in[9];
  const float* cw  = (const float*)d_in[10];
  const float* cb  = (const float*)d_in[11];
  float* out = (float*)d_out;

  char* ws = (char*)d_ws;
  size_t off = 0;
  auto alloc = [&](size_t bytes) {
    void* p = ws + off;
    off = (off + bytes + 255) & ~(size_t)255;
    return p;
  };
  _Float16* Af   = (_Float16*)alloc((size_t)NP * ND * 2);
  _Float16* Of   = (_Float16*)alloc((size_t)NP * ND * 2);
  _Float16* Wqh  = (_Float16*)alloc((size_t)ND * ND * 2);
  _Float16* Wkh  = (_Float16*)alloc((size_t)ND * ND * 2);
  _Float16* Qh   = (_Float16*)alloc((size_t)NP * ND * 2);
  _Float16* Kh   = (_Float16*)alloc((size_t)NP * ND * 2);
  float*    wvc  = (float*)alloc(ND * 4);
  float*    bcst = (float*)alloc(256);
  float*    vsc  = (float*)alloc(NP * 4);
  (void)ws_size; (void)in_sizes; (void)n_in; (void)out_size;

  prep_kernel <<<dim3(NP / 32, ND / 32), dim3(32, 8), 0, stream>>>(rf, sig, pe, Af, Of);
  cvt_w_kernel<<<(2 * ND * ND) / 256, 256, 0, stream>>>(Wq, Wk, Wqh, Wkh);
  wvc_kernel  <<<ND / 256, 256, 0, stream>>>(Wv, cw, bv, wvc, bcst);
  vscal_kernel<<<NP / 256, 256, 0, stream>>>(rf, wvc, bcst, vsc);
  proj_kernel <<<dim3(NP / 16, ND / 64), 128, 0, stream>>>(Af, Wqh, bq, Qh);
  proj_kernel <<<dim3(NP / 16, ND / 64), 128, 0, stream>>>(Of, Wkh, bk, Kh);
  attn_kernel <<<NP / 64, 128, 0, stream>>>(Qh, Kh, vsc, cb, out);
}
